// SimpleGNN_9431748182297
// MI455X (gfx1250) — matrix operand order repaired and checked by reading, not compiled, not run on hardware
//
#include <hip/hip_runtime.h>
#include <hip/hip_fp16.h>

typedef __attribute__((ext_vector_type(16))) _Float16 v16h;
typedef __attribute__((ext_vector_type(8)))  float    v8f;
typedef __attribute__((ext_vector_type(4)))  float    v4f;

#define N_NODES 100000
#define N_EDGES 1600000
#define IN_DIM  128
#define HID     64

// ---------------------------------------------------------------------------
// Degree: atomic count of incoming edges (stored into the dinv buffer)
// ---------------------------------------------------------------------------
__global__ void gcn_deg_kernel(const int* __restrict__ dst, float* __restrict__ deg) {
    int e = blockIdx.x * blockDim.x + threadIdx.x;
    if (e < N_EDGES) atomicAdd(&deg[dst[e]], 1.0f);
}

__global__ void gcn_dinv_kernel(float* __restrict__ deg) {
    int i = blockIdx.x * blockDim.x + threadIdx.x;
    if (i < N_NODES) deg[i] = rsqrtf(deg[i] + 1.0f);
}

// ---------------------------------------------------------------------------
// A-fragment load: 16 rows x 32 K of f32, converted to f16, ISA A-layout:
//   hi==0: halves 0..7 = K kb..kb+7 (kb=k0),   halves 8..15 = K kb+16..kb+23
//   hi==1: same with kb=k0+8  -> covers K k0+8..15 and k0+24..31
// ---------------------------------------------------------------------------
__device__ __forceinline__ v16h load_a_frag(const float* __restrict__ arp,
                                            int kb, bool ok) {
    v4f f0 = {}, f1 = {}, f2 = {}, f3 = {};
    if (ok) {
        const v4f* p = (const v4f*)(arp + kb);
        f0 = p[0]; f1 = p[1];                 // kb .. kb+7
        const v4f* q = (const v4f*)(arp + kb + 16);
        f2 = q[0]; f3 = q[1];                 // kb+16 .. kb+23
    }
    v16h a;
#pragma unroll
    for (int j = 0; j < 4; ++j) {
        a[j]      = (_Float16)f0[j];
        a[j + 4]  = (_Float16)f1[j];
        a[j + 8]  = (_Float16)f2[j];
        a[j + 12] = (_Float16)f3[j];
    }
    return a;
}

// ---------------------------------------------------------------------------
// GEMM: Out[M,64] = A[M,K] @ W[K,64]  via v_wmma_f32_16x16x32_f16.
// W is staged in LDS PRE-SWIZZLED into the exact B-fragment register layout,
// so each lane's 16-half B operand is one contiguous 32-byte LDS read.
// Each wave computes a 32x64 tile (2 M-tiles) so every B fragment read from
// LDS feeds two WMMAs.
// ---------------------------------------------------------------------------
template<int K>
__global__ __launch_bounds__(256) void gcn_gemm_wmma(
    const float* __restrict__ A, const float* __restrict__ W,
    float* __restrict__ Out, int M)
{
    constexpr int KSTEPS = K / 32;
    constexpr int NFRAG  = KSTEPS * 4;                 // 4 N-tiles per k-step
    __shared__ __align__(32) _Float16 sB[NFRAG * 32 * 16];

    // Stage: one (fragment, lane) pair per iteration; gather the 16 strided
    // W values this lane owns in the B layout (lanes 0-15: K k0..k0+7, k0+16..k0+23 at
    // column n0+lane; lanes 16-31: K k0+8..k0+15, k0+24..k0+31) and store contiguously.
    for (int p = threadIdx.x; p < NFRAG * 32; p += 256) {
        const int frag = p >> 5, ln = p & 31;
        const int ks = frag >> 2, nt = frag & 3;
        const int fhi = ln >> 4, flr = ln & 15;
        const float* wp = W + (ks * 32 + fhi * 8) * HID + nt * 16 + flr;
        v16h b;
#pragma unroll
        for (int j = 0; j < 16; ++j) b[j] = (_Float16)wp[(j + (j >> 3) * 8) * HID];
        *(v16h*)&sB[p * 16] = b;
    }
    __syncthreads();

    const int wave = threadIdx.x >> 5;
    const int lane = threadIdx.x & 31;
    const int lr   = lane & 15;
    const int hi   = lane >> 4;
    const long m0  = ((long)blockIdx.x * 8 + wave) * 32;   // 2 tiles per wave
    if (m0 >= M) return;        // wave-uniform: EXEC stays all-1s for WMMA

    const long ar0 = m0 + lr, ar1 = m0 + 16 + lr;
    const bool ok0 = ar0 < M,  ok1 = ar1 < M;
    const float* ap0 = A + ar0 * K;
    const float* ap1 = A + ar1 * K;

    v8f acc[2][4] = {};

#pragma unroll
    for (int ks = 0; ks < KSTEPS; ++ks) {
        const int kb = ks * 32 + hi * 8;
        const v16h a0 = load_a_frag(ap0, kb, ok0);
        const v16h a1 = load_a_frag(ap1, kb, ok1);
#pragma unroll
        for (int nt = 0; nt < 4; ++nt) {
            const v16h b = *(const v16h*)&sB[((ks * 4 + nt) * 32 + lane) * 16];
            acc[0][nt] = __builtin_amdgcn_wmma_f32_16x16x32_f16(
                false, a0, false, b, (short)0, acc[0][nt], false, false);
            acc[1][nt] = __builtin_amdgcn_wmma_f32_16x16x32_f16(
                false, a1, false, b, (short)0, acc[1][nt], false, false);
        }
    }

    // C/D layout: VGPR r, lane<16 -> (M=r, N=lane); lane>=16 -> (M=r+8, N=lane-16)
#pragma unroll
    for (int t = 0; t < 2; ++t) {
#pragma unroll
        for (int r = 0; r < 8; ++r) {
            const long row = m0 + t * 16 + hi * 8 + r;
            if (row < M) {
                float* op = Out + row * HID + lr;
                op[0]  = acc[t][0][r];
                op[16] = acc[t][1][r];
                op[32] = acc[t][2][r];
                op[48] = acc[t][3][r];
            }
        }
    }
}

// ---------------------------------------------------------------------------
// agg[i,:] = xw[i,:] * dinv[i]^2   (self-loop seed; also initializes agg)
// ---------------------------------------------------------------------------
__global__ void gcn_selfloop_kernel(const float* __restrict__ xw,
                                    const float* __restrict__ dinv,
                                    float* __restrict__ agg) {
    long i = (long)blockIdx.x * blockDim.x + threadIdx.x;
    if (i < (long)N_NODES * HID) {
        float d = dinv[i >> 6];
        agg[i] = xw[i] * d * d;
    }
}

// ---------------------------------------------------------------------------
// Edge scatter-add: agg[dst] += xw[src] * dinv[src]*dinv[dst]
// 64 consecutive threads share an edge -> coalesced row reads; L2-resident.
// ---------------------------------------------------------------------------
__global__ void gcn_edge_kernel(const int* __restrict__ src,
                                const int* __restrict__ dst,
                                const float* __restrict__ xw,
                                const float* __restrict__ dinv,
                                float* __restrict__ agg) {
    long g = (long)blockIdx.x * blockDim.x + threadIdx.x;
    if (g >= (long)N_EDGES * HID) return;
    long e = g >> 6;
    int  c = (int)(g & 63);
    int  s = src[e], d = dst[e];
    float coef = dinv[s] * dinv[d];
    atomicAdd(&agg[(long)d * HID + c], xw[(long)s * HID + c] * coef);
}

__global__ void gcn_bias_relu_kernel(const float* __restrict__ agg,
                                     const float* __restrict__ b,
                                     float* __restrict__ h) {
    long i = (long)blockIdx.x * blockDim.x + threadIdx.x;
    if (i < (long)N_NODES * HID) {
        float v = agg[i] + b[i & 63];
        h[i] = v > 0.f ? v : 0.f;
    }
}

// ---------------------------------------------------------------------------
// Column sums of h into gsum[64] (block-local reduce, then 64 atomics/block)
// ---------------------------------------------------------------------------
__global__ void gcn_colsum_kernel(const float* __restrict__ h, float* __restrict__ gsum) {
    __shared__ float sh[256];
    const int t  = threadIdx.x;
    const int c  = t & 63;
    const int rg = t >> 6;   // 0..3
    float s = 0.f;
    for (long r = (long)blockIdx.x * 4 + rg; r < N_NODES; r += (long)gridDim.x * 4)
        s += h[r * HID + c];
    sh[t] = s;
    __syncthreads();
    if (rg == 0) {
        s = sh[c] + sh[c + 64] + sh[c + 128] + sh[c + 192];
        atomicAdd(&gsum[c], s);
    }
}

__global__ void gcn_final_kernel(const float* __restrict__ gsum,
                                 const float* __restrict__ Wf,
                                 const float* __restrict__ bf,
                                 float* __restrict__ out) {
    __shared__ float sh[64];
    int t = threadIdx.x;
    sh[t] = (gsum[t] / (float)N_NODES) * Wf[t];
    __syncthreads();
    if (t < 32) sh[t] += sh[t + 32];
    __syncthreads();
    if (t == 0) {
        float s = 0.f;
        for (int i = 0; i < 32; ++i) s += sh[i];
        s += bf[0];
        out[0] = 1.0f / (1.0f + expf(-s));
    }
}

// ---------------------------------------------------------------------------
extern "C" void kernel_launch(void* const* d_in, const int* in_sizes, int n_in,
                              void* d_out, int out_size, void* d_ws, size_t ws_size,
                              hipStream_t stream) {
    const float* x  = (const float*)d_in[0];
    const int*   ei = (const int*)  d_in[1];
    const float* W1 = (const float*)d_in[2];
    const float* b1 = (const float*)d_in[3];
    const float* W2 = (const float*)d_in[4];
    const float* b2 = (const float*)d_in[5];
    const float* Wf = (const float*)d_in[6];
    const float* bf = (const float*)d_in[7];
    float* out = (float*)d_out;

    const int* src = ei;
    const int* dst = ei + N_EDGES;

    char* ws = (char*)d_ws;
    float* dinv = (float*)ws;                                // 400000 B
    float* gsum = (float*)(ws + 400384);                     // 64 floats
    float* bufA = (float*)(ws + ((size_t)1  << 20));         // 25.6 MB
    float* bufB = (float*)(ws + ((size_t)28 << 20));         // 25.6 MB
    float* bufC = (float*)(ws + ((size_t)55 << 20));         // 25.6 MB

    const long NH = (long)N_NODES * HID;
    const long EH = (long)N_EDGES * HID;
    const int  mtiles32    = (N_NODES + 31) / 32;            // 32 rows per wave
    const int  gemm_blocks = (mtiles32 + 7) / 8;             // 8 waves per block
    const int  nh_blocks   = (int)((NH + 255) / 256);
    const int  eh_blocks   = (int)((EH + 255) / 256);

    // Normalization
    hipMemsetAsync(dinv, 0, N_NODES * sizeof(float), stream);
    hipMemsetAsync(gsum, 0, HID * sizeof(float), stream);
    gcn_deg_kernel<<<(N_EDGES + 255) / 256, 256, 0, stream>>>(dst, dinv);
    gcn_dinv_kernel<<<(N_NODES + 255) / 256, 256, 0, stream>>>(dinv);

    // Layer 1: xw1 = x @ W1 (WMMA); agg = A_norm xw1; h1 = relu(agg + b1)
    gcn_gemm_wmma<IN_DIM><<<gemm_blocks, 256, 0, stream>>>(x, W1, bufA, N_NODES);
    gcn_selfloop_kernel<<<nh_blocks, 256, 0, stream>>>(bufA, dinv, bufB);
    gcn_edge_kernel<<<eh_blocks, 256, 0, stream>>>(src, dst, bufA, dinv, bufB);
    gcn_bias_relu_kernel<<<nh_blocks, 256, 0, stream>>>(bufB, b1, bufC);

    // Layer 2: xw2 = h1 @ W2 (WMMA); agg; h2 = relu(agg + b2)
    gcn_gemm_wmma<HID><<<gemm_blocks, 256, 0, stream>>>(bufC, W2, bufA, N_NODES);
    gcn_selfloop_kernel<<<nh_blocks, 256, 0, stream>>>(bufA, dinv, bufB);
    gcn_edge_kernel<<<eh_blocks, 256, 0, stream>>>(src, dst, bufA, dinv, bufB);
    gcn_bias_relu_kernel<<<nh_blocks, 256, 0, stream>>>(bufB, b2, bufC);

    // Readout: sigmoid(mean(h2) @ Wf + bf)
    gcn_colsum_kernel<<<512, 256, 0, stream>>>(bufC, gsum);
    gcn_final_kernel<<<1, 64, 0, stream>>>(gsum, Wf, bf, out);
}
